// SwinTransformerBlock_5574867550732
// MI455X (gfx1250) — matrix operand order repaired and checked by reading, not compiled, not run on hardware
//
#include <hip/hip_runtime.h>

// ---------------------------------------------------------------------------
// Swin Transformer block for MI455X (gfx1250, wave32, WMMA bf16)
// B=8, H=W=128, C=256, NH=8, hd=32, WS=8, SS=4, N=64, MLP_H=1024
// ---------------------------------------------------------------------------

typedef __attribute__((ext_vector_type(16))) __bf16 v16bf;
typedef __attribute__((ext_vector_type(8)))  __bf16 v8bf;
typedef __attribute__((ext_vector_type(8)))  float  v8f;

union FragBF {
  v16bf v;
  v8bf  h[2];
};

__device__ __forceinline__ v8f wmma_bf16(const FragBF& a, const FragBF& b, v8f c) {
  return __builtin_amdgcn_wmma_f32_16x16x32_bf16(false, a.v, false, b.v,
                                                 (short)0, c, false, false);
}

// ---------------------------------------------------------------------------
// Weight prep: transpose fp32 [K][N] -> bf16 [N][K] once per call, so GEMM
// B staging becomes pure async b128 copies (no convert / strided loads in the
// hot loop).  dst[n*K + k] = src[k*N + n].
// ---------------------------------------------------------------------------
__global__ void k_prep_w(const float* __restrict__ src, __bf16* __restrict__ dst,
                         int kshift, int N) {
  const size_t j = (size_t)blockIdx.x * 256 + threadIdx.x;  // over N*K
  const int K = 1 << kshift;
  const int n = (int)(j >> kshift);
  const int k = (int)(j & (K - 1));
  dst[j] = (__bf16)src[(size_t)k * N + n];
}

// ---------------------------------------------------------------------------
// 128x128-per-block WMMA GEMM core, double-buffered LDS, async A+B staging.
// A:  bf16 row-major [M x K].
// Wt: bf16 pre-transposed weights [N x K] (B-fragment friendly).
// Results delivered per-element via storeD(row, col, val).
// 8 waves: wave wv -> row tiles 2*(wv>>1), 2*(wv>>1)+1; col tiles (wv&1)*4..+3
// => each wave owns a 32x64 output patch (8 WMMA tiles, 64 acc VGPRs).
// ---------------------------------------------------------------------------
template <class StoreD>
__device__ __forceinline__ void gemm_core(const __bf16* __restrict__ A, int lda,
                                          const __bf16* __restrict__ Wt, int K,
                                          int rowBase, int colBase, StoreD storeD) {
  __shared__ __bf16 sA[2][128 * 32];   // A block, row-major [m][k]
  __shared__ __bf16 sB[2][128 * 32];   // B block transposed [n][k]

  const int tid  = threadIdx.x;
  const int lane = tid & 31;
  const int wv   = tid >> 5;
  const int rg   = (wv >> 1) * 2;      // first row tile (0,2,4,6)
  const int cg   = (wv & 1) * 4;       // first col tile (0 or 4)
  const int hi   = (lane >> 4) & 1;    // wave half

  v8f acc[2][4] = {};

  const int r2  = tid >> 1;            // 0..127 (A row / B n-row)
  const int c16 = (tid & 1) * 16;      // 0 or 16 halves
  const int nk  = K >> 5;

  const __bf16* gA0 = &A[(size_t)(rowBase + r2) * lda + c16];
  const __bf16* gB0 = &Wt[(size_t)(colBase + r2) * K + c16];
  unsigned lA[2], lB[2];
  lA[0] = (unsigned)(uintptr_t)&sA[0][r2 * 32 + c16];
  lA[1] = (unsigned)(uintptr_t)&sA[1][r2 * 32 + c16];
  lB[0] = (unsigned)(uintptr_t)&sB[0][r2 * 32 + c16];
  lB[1] = (unsigned)(uintptr_t)&sB[1][r2 * 32 + c16];

  auto stage = [&](int step, int buf) {
    const int k0 = step << 5;
    const __bf16* a = gA0 + k0;
    const __bf16* b = gB0 + k0;
    asm volatile(
        "global_load_async_to_lds_b128 %0, %2, off\n\t"
        "global_load_async_to_lds_b128 %1, %3, off"
        :
        : "v"(lA[buf]), "v"(lA[buf] + 16u), "v"(a), "v"(a + 8)
        : "memory");
    asm volatile(
        "global_load_async_to_lds_b128 %0, %2, off\n\t"
        "global_load_async_to_lds_b128 %1, %3, off"
        :
        : "v"(lB[buf]), "v"(lB[buf] + 16u), "v"(b), "v"(b + 8)
        : "memory");
  };

  stage(0, 0);
  asm volatile("s_wait_asynccnt 0" ::: "memory");
  __syncthreads();

  for (int step = 0; step < nk; ++step) {
    const int buf = step & 1;
    if (step + 1 < nk) stage(step + 1, buf ^ 1);

    // A fragments: row m; halves 0..7 -> k = hi*8 + 0..7,
    // halves 8..15 -> k = 16 + hi*8 + 0..7 (ISA 16-bit A layout).
    FragBF a[2];
#pragma unroll
    for (int r = 0; r < 2; ++r) {
      const int m = (rg + r) * 16 + (lane & 15);
      a[r].h[0] = *(const v8bf*)&sA[buf][m * 32 + hi * 8];
      a[r].h[1] = *(const v8bf*)&sA[buf][m * 32 + 16 + hi * 8];
    }
#pragma unroll
    for (int ct = 0; ct < 4; ++ct) {
      // B fragment: col n = tile*16 + (lane&15); same k order as A.
      FragBF b;
      const int n = (cg + ct) * 16 + (lane & 15);
      const __bf16* bp = &sB[buf][n * 32 + hi * 8];
      b.h[0] = *(const v8bf*)bp;
      b.h[1] = *(const v8bf*)(bp + 16);
#pragma unroll
      for (int r = 0; r < 2; ++r) acc[r][ct] = wmma_bf16(a[r], b, acc[r][ct]);
    }

    asm volatile("s_wait_asynccnt 0" ::: "memory");
    __syncthreads();
  }

  // D layout: element v of lane -> row = v + 8*hi, col = lane&15.
#pragma unroll
  for (int r = 0; r < 2; ++r)
#pragma unroll
    for (int ct = 0; ct < 4; ++ct)
#pragma unroll
      for (int v = 0; v < 8; ++v) {
        const int row = rowBase + (rg + r) * 16 + v + hi * 8;
        const int col = colBase + (cg + ct) * 16 + (lane & 15);
        storeD(row, col, acc[r][ct][v]);
      }
}

// ---------------------------------------------------------------------------
// LayerNorm 1 + cyclic shift (-4,-4) + window partition -> bf16 [2048*64, 256]
// One wave per token.
// ---------------------------------------------------------------------------
__global__ void k_ln1_shift(const float* __restrict__ x, const float* __restrict__ g,
                            const float* __restrict__ beta, __bf16* __restrict__ hw) {
  const int lane = threadIdx.x & 31;
  const int wv   = threadIdx.x >> 5;
  const int tok  = blockIdx.x * 8 + wv;  // 0..131071
  const float* xp = x + (size_t)tok * 256;

  float vals[8];
  *(float4*)&vals[0] = ((const float4*)xp)[lane * 2];
  *(float4*)&vals[4] = ((const float4*)xp)[lane * 2 + 1];

  float s1 = 0.f, s2 = 0.f;
#pragma unroll
  for (int i = 0; i < 8; ++i) { s1 += vals[i]; s2 += vals[i] * vals[i]; }
#pragma unroll
  for (int m = 1; m < 32; m <<= 1) {
    s1 += __shfl_xor(s1, m, 32);
    s2 += __shfl_xor(s2, m, 32);
  }
  const float mean = s1 * (1.f / 256.f);
  const float var  = s2 * (1.f / 256.f) - mean * mean;
  const float rstd = rsqrtf(var + 1e-5f);

  const int b  = tok >> 14;
  const int hh = (tok >> 7) & 127;
  const int ww = tok & 127;
  const int hs  = (hh - 4) & 127;  // rolled position
  const int ws2 = (ww - 4) & 127;
  const int win = b * 256 + (hs >> 3) * 16 + (ws2 >> 3);
  const int t   = (hs & 7) * 8 + (ws2 & 7);

  union { v8bf v; __bf16 e[8]; } o;
#pragma unroll
  for (int i = 0; i < 8; ++i) {
    const int c = lane * 8 + i;
    o.e[i] = (__bf16)((vals[i] - mean) * rstd * g[c] + beta[c]);
  }
  *(v8bf*)&hw[((size_t)win * 64 + t) * 256 + lane * 8] = o.v;
}

// LayerNorm 2 (no shift): fp32 in -> bf16 out, same row index.
__global__ void k_ln2(const float* __restrict__ x, const float* __restrict__ g,
                      const float* __restrict__ beta, __bf16* __restrict__ out) {
  const int lane = threadIdx.x & 31;
  const int wv   = threadIdx.x >> 5;
  const int tok  = blockIdx.x * 8 + wv;
  const float* xp = x + (size_t)tok * 256;

  float vals[8];
  *(float4*)&vals[0] = ((const float4*)xp)[lane * 2];
  *(float4*)&vals[4] = ((const float4*)xp)[lane * 2 + 1];

  float s1 = 0.f, s2 = 0.f;
#pragma unroll
  for (int i = 0; i < 8; ++i) { s1 += vals[i]; s2 += vals[i] * vals[i]; }
#pragma unroll
  for (int m = 1; m < 32; m <<= 1) {
    s1 += __shfl_xor(s1, m, 32);
    s2 += __shfl_xor(s2, m, 32);
  }
  const float mean = s1 * (1.f / 256.f);
  const float var  = s2 * (1.f / 256.f) - mean * mean;
  const float rstd = rsqrtf(var + 1e-5f);

  union { v8bf v; __bf16 e[8]; } o;
#pragma unroll
  for (int i = 0; i < 8; ++i) {
    const int c = lane * 8 + i;
    o.e[i] = (__bf16)((vals[i] - mean) * rstd * g[c] + beta[c]);
  }
  *(v8bf*)&out[(size_t)tok * 256 + lane * 8] = o.v;
}

// ---------------------------------------------------------------------------
// Fused QKV GEMM: hw[131072x256] @ WqkvT[768x256]^T (+bias).
// Q gets 1/sqrt(32) baked in; V is stored transposed (d-major) for attention.
// Destination region (q/k/v) is block-uniform: colBase>>8.
// ---------------------------------------------------------------------------
__global__ void k_gemm_qkv(const __bf16* __restrict__ hw, const __bf16* __restrict__ wt,
                           const float* __restrict__ qb, const float* __restrict__ kvb,
                           __bf16* __restrict__ q, __bf16* __restrict__ k,
                           __bf16* __restrict__ vt) {
  const int rowBase = blockIdx.x * 128;
  const int colBase = blockIdx.y * 128;
  const int region  = colBase >> 8;   // 0 = q, 1 = k, 2 = v (uniform per block)
  auto storeD = [&](int r, int n, float val) {
    const int w = r >> 6, t = r & 63;
    if (region == 0) {
      const int hd = n >> 5, d = n & 31;
      const float o = (val + qb[n]) * 0.17677669529663687f;  // 1/sqrt(32)
      q[((size_t)(w * 8 + hd) * 64 + t) * 32 + d] = (__bf16)o;
    } else if (region == 1) {
      const int np = n - 256;
      const int hd = np >> 5, d = np & 31;
      k[((size_t)(w * 8 + hd) * 64 + t) * 32 + d] = (__bf16)(val + kvb[np]);
    } else {
      const int np = n - 256;             // kvb index
      const int nq = n - 512;             // v feature
      const int hd = nq >> 5, d = nq & 31;
      vt[((size_t)(w * 8 + hd) * 32 + d) * 64 + t] = (__bf16)(val + kvb[np]);
    }
  };
  gemm_core(hw, 256, wt, 256, rowBase, colBase, storeD);
}

// ---------------------------------------------------------------------------
// Windowed attention: one block per window, one wave per head.
// S = Q K^T (+rel-pos bias +shift mask), softmax, O = P V.
// ---------------------------------------------------------------------------
__device__ __forceinline__ int swin_region(int p) {
  return (p < 120) ? 0 : ((p < 124) ? 1 : 2);
}

__global__ void k_attn(const __bf16* __restrict__ q, const __bf16* __restrict__ k,
                       const __bf16* __restrict__ vt, const float* __restrict__ rpb,
                       __bf16* __restrict__ ao) {
  __shared__ __bf16 sP[8][64 * 64];  // per-wave softmax probs, 64 KB total

  const int lane = threadIdx.x & 31;
  const int hd   = threadIdx.x >> 5;   // head = wave
  const int hi   = (lane >> 4) & 1;
  const int win  = blockIdx.x;
  const int wi   = win & 255;          // window index within image

  const __bf16* qp = q  + ((size_t)(win * 8 + hd)) * 64 * 32;
  const __bf16* kp = k  + ((size_t)(win * 8 + hd)) * 64 * 32;
  const __bf16* vp = vt + ((size_t)(win * 8 + hd)) * 32 * 64;

  // Q fragments for the 4 query row-tiles (A layout, K=32 contiguous rows).
  FragBF qa[4];
#pragma unroll
  for (int mi = 0; mi < 4; ++mi) {
    const int m = mi * 16 + (lane & 15);
    qa[mi].h[0] = *(const v8bf*)&qp[m * 32 + hi * 8];
    qa[mi].h[1] = *(const v8bf*)&qp[m * 32 + 16 + hi * 8];
  }
  // K fragments (B layout): column = key token, elements run over d.
  FragBF kf[4];
#pragma unroll
  for (int ni = 0; ni < 4; ++ni) {
    const int n = ni * 16 + (lane & 15);
    const __bf16* p = &kp[n * 32 + hi * 8];
    kf[ni].h[0] = *(const v8bf*)p;
    kf[ni].h[1] = *(const v8bf*)(p + 16);
  }

  v8f s[4][4] = {};
#pragma unroll
  for (int mi = 0; mi < 4; ++mi)
#pragma unroll
    for (int ni = 0; ni < 4; ++ni)
      s[mi][ni] = wmma_bf16(qa[mi], kf[ni], s[mi][ni]);

  // Relative-position bias + shift mask, computed in D layout.
  const int whi = wi >> 4, wwi = wi & 15;
#pragma unroll
  for (int mi = 0; mi < 4; ++mi)
#pragma unroll
    for (int ni = 0; ni < 4; ++ni)
#pragma unroll
      for (int v = 0; v < 8; ++v) {
        const int t1 = mi * 16 + v + hi * 8;      // query token
        const int t2 = ni * 16 + (lane & 15);     // key token
        const int r1i = t1 >> 3, r1j = t1 & 7;
        const int r2i = t2 >> 3, r2j = t2 & 7;
        const int idx = (r1i - r2i + 7) * 15 + (r1j - r2j + 7);
        const float bias = rpb[idx * 8 + hd];
        const int l1 = swin_region(whi * 8 + r1i) * 3 + swin_region(wwi * 8 + r1j);
        const int l2 = swin_region(whi * 8 + r2i) * 3 + swin_region(wwi * 8 + r2j);
        s[mi][ni][v] += bias + ((l1 == l2) ? 0.f : -100.f);
      }

  // Row softmax: reduce across the 16-lane column group and 4 column tiles.
#pragma unroll
  for (int mi = 0; mi < 4; ++mi)
#pragma unroll
    for (int v = 0; v < 8; ++v) {
      float mx = s[mi][0][v];
#pragma unroll
      for (int ni = 1; ni < 4; ++ni) mx = fmaxf(mx, s[mi][ni][v]);
#pragma unroll
      for (int msk = 1; msk < 16; msk <<= 1) mx = fmaxf(mx, __shfl_xor(mx, msk, 32));
      float sum = 0.f;
      float e[4];
#pragma unroll
      for (int ni = 0; ni < 4; ++ni) { e[ni] = __expf(s[mi][ni][v] - mx); sum += e[ni]; }
#pragma unroll
      for (int msk = 1; msk < 16; msk <<= 1) sum += __shfl_xor(sum, msk, 32);
      const float inv = 1.f / sum;
      const int t1 = mi * 16 + v + hi * 8;
#pragma unroll
      for (int ni = 0; ni < 4; ++ni)
        sP[hd][t1 * 64 + ni * 16 + (lane & 15)] = (__bf16)(e[ni] * inv);
    }
  __syncthreads();

  // O = P @ V  (K=64 -> two k-steps; V pre-transposed so B loads are contiguous)
#pragma unroll
  for (int mi = 0; mi < 4; ++mi) {
    v8f o[2] = {};
#pragma unroll
    for (int kk = 0; kk < 2; ++kk) {
      FragBF pa;
      const int m = mi * 16 + (lane & 15);
      const __bf16* pp = &sP[hd][m * 64 + kk * 32];
      pa.h[0] = *(const v8bf*)&pp[hi * 8];
      pa.h[1] = *(const v8bf*)&pp[16 + hi * 8];
#pragma unroll
      for (int di = 0; di < 2; ++di) {
        FragBF vb;
        const int d = di * 16 + (lane & 15);
        const __bf16* vv = &vp[d * 64 + kk * 32 + hi * 8];
        vb.h[0] = *(const v8bf*)vv;
        vb.h[1] = *(const v8bf*)(vv + 16);
        o[di] = wmma_bf16(pa, vb, o[di]);
      }
    }
#pragma unroll
    for (int di = 0; di < 2; ++di)
#pragma unroll
      for (int v = 0; v < 8; ++v) {
        const int t = mi * 16 + v + hi * 8;
        const int d = di * 16 + (lane & 15);
        ao[((size_t)win * 64 + t) * 256 + hd * 32 + d] = (__bf16)o[di][v];
      }
  }
}

// ---------------------------------------------------------------------------
// Projection GEMM + window reverse + reverse shift + residual -> fp32 x1.
// ---------------------------------------------------------------------------
__global__ void k_gemm_proj(const __bf16* __restrict__ ao, const __bf16* __restrict__ wt,
                            const float* __restrict__ pb, const float* __restrict__ x0,
                            float* __restrict__ x1) {
  const int rowBase = blockIdx.x * 128;
  const int colBase = blockIdx.y * 128;
  auto storeD = [&](int r, int n, float val) {
    const int w = r >> 6, t = r & 63;
    const int b = w >> 8, wi = w & 255;
    const int hs  = (wi >> 4) * 8 + (t >> 3);
    const int wsp = (wi & 15) * 8 + (t & 7);
    const int hh = (hs + 4) & 127;   // reverse roll
    const int ww = (wsp + 4) & 127;
    const size_t tok = (size_t)b * 16384 + hh * 128 + ww;
    x1[tok * 256 + n] = val + pb[n] + x0[tok * 256 + n];
  };
  gemm_core(ao, 256, wt, 256, rowBase, colBase, storeD);
}

// ---------------------------------------------------------------------------
// MLP fc1 + exact GELU -> bf16 hidden [131072 x 1024]
// ---------------------------------------------------------------------------
__global__ void k_gemm_fc1(const __bf16* __restrict__ a, const __bf16* __restrict__ wt,
                           const float* __restrict__ bias, __bf16* __restrict__ h) {
  const int rowBase = blockIdx.x * 128;
  const int colBase = blockIdx.y * 128;
  auto storeD = [&](int r, int n, float val) {
    const float z = val + bias[n];
    const float g = 0.5f * z * (1.f + erff(z * 0.70710678118654752f));
    h[(size_t)r * 1024 + n] = (__bf16)g;
  };
  gemm_core(a, 256, wt, 256, rowBase, colBase, storeD);
}

// ---------------------------------------------------------------------------
// MLP fc2 + bias + residual -> fp32 output
// ---------------------------------------------------------------------------
__global__ void k_gemm_fc2(const __bf16* __restrict__ h, const __bf16* __restrict__ wt,
                           const float* __restrict__ bias, const float* __restrict__ x1,
                           float* __restrict__ out) {
  const int rowBase = blockIdx.x * 128;
  const int colBase = blockIdx.y * 128;
  auto storeD = [&](int r, int n, float val) {
    out[(size_t)r * 256 + n] = val + bias[n] + x1[(size_t)r * 256 + n];
  };
  gemm_core(h, 1024, wt, 1024, rowBase, colBase, storeD);
}

// ---------------------------------------------------------------------------
// Launch
// ---------------------------------------------------------------------------
extern "C" void kernel_launch(void* const* d_in, const int* in_sizes, int n_in,
                              void* d_out, int out_size, void* d_ws, size_t ws_size,
                              hipStream_t stream) {
  const float* x     = (const float*)d_in[0];
  const float* g1    = (const float*)d_in[1];
  const float* b1    = (const float*)d_in[2];
  const float* qw    = (const float*)d_in[3];
  const float* qb    = (const float*)d_in[4];
  const float* kvw   = (const float*)d_in[5];
  const float* kvb   = (const float*)d_in[6];
  const float* rpb   = (const float*)d_in[7];
  const float* projw = (const float*)d_in[8];
  const float* projb = (const float*)d_in[9];
  const float* g2    = (const float*)d_in[10];
  const float* b2    = (const float*)d_in[11];
  const float* fc1w  = (const float*)d_in[12];
  const float* fc1b  = (const float*)d_in[13];
  const float* fc2w  = (const float*)d_in[14];
  const float* fc2b  = (const float*)d_in[15];

  char* ws = (char*)d_ws;
  const size_t SZ = 131072ull * 256ull * 2ull;          // 64 MiB bf16 activation
  __bf16* hw    = (__bf16*)(ws + 0);                    // LN1 out; reused as attn out
  __bf16* qbuf  = (__bf16*)(ws + SZ);
  __bf16* kbuf  = (__bf16*)(ws + 2 * SZ);
  __bf16* vtbuf = (__bf16*)(ws + 3 * SZ);
  float*  x1    = (float*)(ws + 4 * SZ);                // fp32 residual (128 MiB)
  __bf16* ln2o  = (__bf16*)(ws + 4 * SZ + 131072ull * 256ull * 4ull);
  __bf16* mlp1  = (__bf16*)(ws + 0);                    // reuses hw/q/k/vt (256 MiB)

  // Pre-transposed bf16 weights ([N][K]) after ln2o.
  char* wbase = ws + 4 * SZ + 131072ull * 256ull * 4ull + SZ;
  __bf16* wqkvT = (__bf16*)(wbase);                     // 768x256
  __bf16* wprojT = (__bf16*)(wbase + 393216);           // 256x256
  __bf16* wfc1T  = (__bf16*)(wbase + 393216 + 131072);  // 1024x256
  __bf16* wfc2T  = (__bf16*)(wbase + 393216 + 131072 + 524288);  // 256x1024

  float* out = (float*)d_out;

  // Weight prep (tiny, ~1.5 MiB of bf16 written once per call).
  k_prep_w<<<256, 256, 0, stream>>>(qw, wqkvT, 8, 256);                 // q rows
  k_prep_w<<<512, 256, 0, stream>>>(kvw, wqkvT + 256 * 256, 8, 512);    // kv rows
  k_prep_w<<<256, 256, 0, stream>>>(projw, wprojT, 8, 256);
  k_prep_w<<<1024, 256, 0, stream>>>(fc1w, wfc1T, 8, 1024);
  k_prep_w<<<1024, 256, 0, stream>>>(fc2w, wfc2T, 10, 256);

  k_ln1_shift<<<16384, 256, 0, stream>>>(x, g1, b1, hw);
  k_gemm_qkv<<<dim3(1024, 6), 256, 0, stream>>>(hw, wqkvT, qb, kvb,
                                                qbuf, kbuf, vtbuf);
  k_attn<<<2048, 256, 0, stream>>>(qbuf, kbuf, vtbuf, rpb, hw);
  k_gemm_proj<<<dim3(1024, 2), 256, 0, stream>>>(hw, wprojT, projb, x, x1);
  k_ln2<<<16384, 256, 0, stream>>>(x1, g2, b2, ln2o);
  k_gemm_fc1<<<dim3(1024, 8), 256, 0, stream>>>(ln2o, wfc1T, fc1b, mlp1);
  k_gemm_fc2<<<dim3(1024, 2), 256, 0, stream>>>(mlp1, wfc2T, fc2b, x1, out);
}
